// HeterogeneousGraphTransformer_15367392985644
// MI455X (gfx1250) — matrix operand order repaired and checked by reading, not compiled, not run on hardware
//
#include <hip/hip_runtime.h>
#include <hip/hip_bf16.h>

// ============================================================================
// HGT forward for MI455X (gfx1250, wave32, WMMA).
//  - 3 projections per relation fused as one GEMM: S(N x 256) @ Wcat_r^T (256 x 768),
//    computed with V_WMMA_F32_16X16X32_BF16 using a hi/lo bf16 split of both
//    operands (3 WMMAs per k-step) for ~fp32 accuracy on the bf16 matrix pipe.
//  - Edge phase is restructured to be atomic-free for floats: per-relation CSR
//    build, then one wave per receiver node does max / sum-exp / weighted-V
//    accumulation with coalesced 1KB row reads (Q/K/V stay L2-resident:
//    153 MB < 192 MB L2).
// ============================================================================

typedef __bf16 bf16;
typedef __attribute__((ext_vector_type(16))) __bf16 v16bf;
typedef __attribute__((ext_vector_type(8)))  __bf16 v8bf;
typedef __attribute__((ext_vector_type(8)))  float  v8f;

#define DIM  256
#define COLS 768   // V | Q | K concatenated per relation

// ---------------------------------------------------------------------------
// fp32 -> (bf16 hi, bf16 lo): hi is truncation (exact high bits), lo is the
// rounded residual. a*b ~= hi_a*hi_b + hi_a*lo_b + lo_a*hi_b to ~2^-16 rel.
// ---------------------------------------------------------------------------
__device__ inline void split_store(float v, unsigned short* hi, unsigned short* lo, int i) {
  unsigned u  = __float_as_uint(v);
  unsigned hu = u & 0xFFFF0000u;
  hi[i] = (unsigned short)(hu >> 16);
  float lf = v - __uint_as_float(hu);
  unsigned lu = __float_as_uint(lf);
  unsigned r  = (lu + 0x7FFFu + ((lu >> 16) & 1u)) >> 16;   // RNE to bf16
  lo[i] = (unsigned short)r;
}

__global__ void split_s_kernel(const float* __restrict__ x,
                               unsigned short* __restrict__ hi,
                               unsigned short* __restrict__ lo, int n) {
  int i = blockIdx.x * blockDim.x + threadIdx.x;
  if (i < n) split_store(x[i], hi, lo, i);
}

// Build Wcat (2304 x 256): per relation r, rows [0,256)=W[r], [256,512)=WQ[r],
// [512,768)=WK[r].  Row j of Wcat is column j of the GEMM's B operand.
__global__ void build_w_kernel(const float* __restrict__ W,
                               const float* __restrict__ WQ,
                               const float* __restrict__ WK,
                               unsigned short* __restrict__ hi,
                               unsigned short* __restrict__ lo, int n) {
  int i = blockIdx.x * blockDim.x + threadIdx.x;
  if (i >= n) return;
  int k    = i & 255;
  int row  = i >> 8;            // 0..2303
  int rel  = row / COLS;
  int sub  = row - rel * COLS;  // 0..767
  int which = sub >> 8;         // 0=V(W), 1=Q(WQ), 2=K(WK)
  int wrow  = sub & 255;
  const float* src = (which == 0) ? W : (which == 1) ? WQ : WK;
  split_store(src[((size_t)rel * DIM + wrow) * DIM + k], hi, lo, i);
}

__global__ void zero_f32_kernel(float* p, int n) {
  int i = blockIdx.x * blockDim.x + threadIdx.x;
  if (i < n) p[i] = 0.0f;
}
__global__ void zero_i32_kernel(int* p, int n) {
  int i = blockIdx.x * blockDim.x + threadIdx.x;
  if (i < n) p[i] = 0;
}

// ---------------------------------------------------------------------------
// GEMM: Out(M x 768) = S(M x 256) @ Wcat_r^T, bf16x3 split via WMMA.
// Block = 128 threads = 4 waves; block tile 64(M) x 64(N); wave tile 16 x 64.
// A-frag (16x32 bf16)  lane L: row L%16, elems j<8 -> K=8h+j, j>=8 -> K=16+8h+(j-8)
// B-frag (32x16 bf16)  lane L: col L%16, elems j<8 -> K=8h+j, j>=8 -> K=16+8h+(j-8)
//                      (B col = Wcat row, K contiguous in memory -> 16B loads)
// C/D (16x16 f32)      lane L: col L%16, VGPR v -> row v + 8*(L/16)
// ---------------------------------------------------------------------------
__global__ __launch_bounds__(128)
void gemm_wmma_kernel(const unsigned short* __restrict__ Shi_,
                      const unsigned short* __restrict__ Slo_,
                      const unsigned short* __restrict__ Bhi_,
                      const unsigned short* __restrict__ Blo_,
                      float* __restrict__ Out, int M) {
  const bf16* Shi = (const bf16*)Shi_;
  const bf16* Slo = (const bf16*)Slo_;
  const bf16* Bhi = (const bf16*)Bhi_;
  const bf16* Blo = (const bf16*)Blo_;

  const int lane = threadIdx.x & 31;
  const int wave = threadIdx.x >> 5;
  const int rowBase = (blockIdx.x * 4 + wave) << 4;
  const int colBase = blockIdx.y << 6;
  const int r = lane & 15;
  const int h = lane >> 4;

  int arow = rowBase + r;
  if (arow >= M) arow = M - 1;                 // clamp loads; stores are guarded
  const bf16* aH = Shi + (size_t)arow * DIM;
  const bf16* aL = Slo + (size_t)arow * DIM;
  const bf16* bH = Bhi + (size_t)(colBase + r) * DIM;
  const bf16* bL = Blo + (size_t)(colBase + r) * DIM;

  v8f acc[4] = {};

  for (int kb = 0; kb < DIM; kb += 32) {
    union { v16bf v; v8bf p[2]; } a_hi, a_lo;
    a_hi.p[0] = *(const v8bf*)(aH + kb + 8 * h);
    a_hi.p[1] = *(const v8bf*)(aH + kb + 16 + 8 * h);
    a_lo.p[0] = *(const v8bf*)(aL + kb + 8 * h);
    a_lo.p[1] = *(const v8bf*)(aL + kb + 16 + 8 * h);
#pragma unroll
    for (int t = 0; t < 4; ++t) {
      union { v16bf v; v8bf p[2]; } b_hi, b_lo;
      const bf16* ph = bH + (size_t)t * 16 * DIM + kb + 8 * h;
      const bf16* pl = bL + (size_t)t * 16 * DIM + kb + 8 * h;
      b_hi.p[0] = *(const v8bf*)(ph);
      b_hi.p[1] = *(const v8bf*)(ph + 16);
      b_lo.p[0] = *(const v8bf*)(pl);
      b_lo.p[1] = *(const v8bf*)(pl + 16);
      acc[t] = __builtin_amdgcn_wmma_f32_16x16x32_bf16(false, a_hi.v, false, b_hi.v,
                                                       (short)0, acc[t], false, false);
      acc[t] = __builtin_amdgcn_wmma_f32_16x16x32_bf16(false, a_hi.v, false, b_lo.v,
                                                       (short)0, acc[t], false, false);
      acc[t] = __builtin_amdgcn_wmma_f32_16x16x32_bf16(false, a_lo.v, false, b_hi.v,
                                                       (short)0, acc[t], false, false);
    }
  }

#pragma unroll
  for (int t = 0; t < 4; ++t) {
    int col = colBase + t * 16 + r;
#pragma unroll
    for (int v = 0; v < 8; ++v) {
      int row = rowBase + v + 8 * h;
      if (row < M) Out[(size_t)row * COLS + col] = acc[t][v];
    }
  }
}

// ---------------------------------------------------------------------------
// One wave per directed edge: score = dot(Q[rec], K[nbr]) / sqrt(64).
// ---------------------------------------------------------------------------
__global__ __launch_bounds__(256)
void edge_score_kernel(const int* __restrict__ edges, int E2, int E1,
                       const float* __restrict__ qkv, float* __restrict__ scores) {
  int wid  = blockIdx.x * 8 + (threadIdx.x >> 5);
  int lane = threadIdx.x & 31;
  if (wid >= E2) return;
  int rec, nbr;
  if (wid < E1) { rec = edges[2 * wid];     nbr = edges[2 * wid + 1]; }
  else          { int i = wid - E1; rec = edges[2 * i + 1]; nbr = edges[2 * i]; }
  const float4* q = (const float4*)(qkv + (size_t)rec * COLS + 256);
  const float4* k = (const float4*)(qkv + (size_t)nbr * COLS + 512);
  float4 q0 = q[2 * lane], q1 = q[2 * lane + 1];
  float4 k0 = k[2 * lane], k1 = k[2 * lane + 1];
  float s = q0.x * k0.x + q0.y * k0.y + q0.z * k0.z + q0.w * k0.w
          + q1.x * k1.x + q1.y * k1.y + q1.z * k1.z + q1.w * k1.w;
#pragma unroll
  for (int m = 16; m; m >>= 1) s += __shfl_xor(s, m, 32);
  if (lane == 0) scores[wid] = s * 0.125f;    // 1/sqrt(d_k), d_k = 64
}

__global__ void degree_kernel(const int* __restrict__ edges, int E2, int E1,
                              int* __restrict__ deg) {
  int e = blockIdx.x * blockDim.x + threadIdx.x;
  if (e >= E2) return;
  int rec = (e < E1) ? edges[2 * e] : edges[2 * (e - E1) + 1];
  atomicAdd(&deg[rec], 1);
}

// Single-block exclusive scan over n degrees (n ~ 50k; trivial cost).
__global__ void exscan_kernel(const int* __restrict__ deg, int* __restrict__ rowstart, int n) {
  __shared__ int buf[1024];
  __shared__ int carry;
  int tid = threadIdx.x;
  if (tid == 0) carry = 0;
  __syncthreads();
  for (int base = 0; base < n; base += 1024) {
    int x = (base + tid < n) ? deg[base + tid] : 0;
    buf[tid] = x;
    __syncthreads();
    for (int off = 1; off < 1024; off <<= 1) {
      int t = (tid >= off) ? buf[tid - off] : 0;
      __syncthreads();
      buf[tid] += t;
      __syncthreads();
    }
    int c = carry;
    if (base + tid < n) rowstart[base + tid] = c + buf[tid] - x;
    __syncthreads();
    if (tid == 0) carry += buf[1023];
    __syncthreads();
  }
  if (tid == 0) rowstart[n] = carry;
}

__global__ void scatter_kernel(const int* __restrict__ edges, int E2, int E1,
                               const int* __restrict__ rowstart, int* __restrict__ cursor,
                               int* __restrict__ csr_nbr, int* __restrict__ csr_eidx) {
  int e = blockIdx.x * blockDim.x + threadIdx.x;
  if (e >= E2) return;
  int rec, nbr;
  if (e < E1) { rec = edges[2 * e];     nbr = edges[2 * e + 1]; }
  else        { int i = e - E1; rec = edges[2 * i + 1]; nbr = edges[2 * i]; }
  int pos = rowstart[rec] + atomicAdd(&cursor[rec], 1);
  csr_nbr[pos]  = nbr;
  csr_eidx[pos] = e;
}

// ---------------------------------------------------------------------------
// One wave per receiver node: segment softmax + weighted V aggregation.
// No float atomics; V rows read as coalesced 1KB bursts (L2-resident).
// ---------------------------------------------------------------------------
__global__ __launch_bounds__(256)
void node_agg_kernel(const int* __restrict__ rowstart, const int* __restrict__ csr_nbr,
                     const int* __restrict__ csr_eidx, const float* __restrict__ scores,
                     const float* __restrict__ qkv, float* __restrict__ out, int n) {
  int wid  = blockIdx.x * 8 + (threadIdx.x >> 5);
  int lane = threadIdx.x & 31;
  if (wid >= n) return;
  int s0 = rowstart[wid], s1 = rowstart[wid + 1];
  if (s0 >= s1) return;                         // isolated node: stays 0

  float m = -3.402823466e38f;
  for (int i = s0 + lane; i < s1; i += 32) m = fmaxf(m, scores[csr_eidx[i]]);
#pragma unroll
  for (int k = 16; k; k >>= 1) m = fmaxf(m, __shfl_xor(m, k, 32));

  float z = 0.0f;
  for (int i = s0 + lane; i < s1; i += 32) z += __expf(scores[csr_eidx[i]] - m);
#pragma unroll
  for (int k = 16; k; k >>= 1) z += __shfl_xor(z, k, 32);
  float inv = 1.0f / z;

  float4 a0 = {0, 0, 0, 0}, a1 = {0, 0, 0, 0};
  for (int i = s0; i < s1; ++i) {               // whole wave per edge: coalesced
    int   nbr = csr_nbr[i];
    float w   = __expf(scores[csr_eidx[i]] - m) * inv;
    const float4* v = (const float4*)(qkv + (size_t)nbr * COLS);   // V at offset 0
    float4 v0 = v[2 * lane], v1 = v[2 * lane + 1];
    a0.x += w * v0.x; a0.y += w * v0.y; a0.z += w * v0.z; a0.w += w * v0.w;
    a1.x += w * v1.x; a1.y += w * v1.y; a1.z += w * v1.z; a1.w += w * v1.w;
  }

  float4* o = (float4*)(out + (size_t)wid * DIM);
  float4 o0 = o[2 * lane], o1 = o[2 * lane + 1];
  o0.x += a0.x; o0.y += a0.y; o0.z += a0.z; o0.w += a0.w;
  o1.x += a1.x; o1.y += a1.y; o1.z += a1.z; o1.w += a1.w;
  o[2 * lane] = o0; o[2 * lane + 1] = o1;
}

__global__ void relu_kernel(float* p, int n) {
  int i = blockIdx.x * blockDim.x + threadIdx.x;
  if (i < n) p[i] = fmaxf(p[i], 0.0f);
}

// ===========================================================================
extern "C" void kernel_launch(void* const* d_in, const int* in_sizes, int n_in,
                              void* d_out, int out_size, void* d_ws, size_t ws_size,
                              hipStream_t stream) {
  const float* S   = (const float*)d_in[0];
  const int*   eT  = (const int*)d_in[1];
  const int*   eC  = (const int*)d_in[2];
  const int*   eS  = (const int*)d_in[3];
  const float* W   = (const float*)d_in[4];
  const float* WQ  = (const float*)d_in[5];
  const float* WK  = (const float*)d_in[6];
  float* out = (float*)d_out;

  const int N  = in_sizes[0] / DIM;   // 50000
  const int E1 = in_sizes[1] / 2;     // 150000 undirected
  const int E2 = 2 * E1;              // directed

  // ---- workspace carve-up (256B aligned slices) ----
  char*  base = (char*)d_ws;
  size_t off  = 0;
  auto take = [&](size_t bytes) -> void* {
    void* p = base + off;
    off = (off + bytes + 255) & ~(size_t)255;
    return p;
  };
  unsigned short* Shi = (unsigned short*)take((size_t)N * DIM * 2);
  unsigned short* Slo = (unsigned short*)take((size_t)N * DIM * 2);
  unsigned short* Whi = (unsigned short*)take((size_t)3 * COLS * DIM * 2);
  unsigned short* Wlo = (unsigned short*)take((size_t)3 * COLS * DIM * 2);
  float* qkv      = (float*)take((size_t)N * COLS * 4);   // per-relation V|Q|K
  float* scores   = (float*)take((size_t)E2 * 4);
  int*   rowstart = (int*)take((size_t)(N + 1) * 4);
  int*   cursor   = (int*)take((size_t)N * 4);            // degree, then cursor
  int*   csr_nbr  = (int*)take((size_t)E2 * 4);
  int*   csr_eidx = (int*)take((size_t)E2 * 4);
  (void)ws_size; (void)n_in; (void)out_size;

  const int nS = N * DIM;
  const int nW = 3 * COLS * DIM;

  split_s_kernel<<<(nS + 255) / 256, 256, 0, stream>>>(S, Shi, Slo, nS);
  build_w_kernel<<<(nW + 255) / 256, 256, 0, stream>>>(W, WQ, WK, Whi, Wlo, nW);
  zero_f32_kernel<<<(nS + 255) / 256, 256, 0, stream>>>(out, nS);

  const int* edge_lists[3] = {eT, eC, eS};
  dim3 gemmGrid((N + 63) / 64, COLS / 64);

  for (int r = 0; r < 3; ++r) {
    const int* edges = edge_lists[r];
    gemm_wmma_kernel<<<gemmGrid, 128, 0, stream>>>(
        Shi, Slo, Whi + (size_t)r * COLS * DIM, Wlo + (size_t)r * COLS * DIM, qkv, N);

    edge_score_kernel<<<(E2 + 7) / 8, 256, 0, stream>>>(edges, E2, E1, qkv, scores);

    zero_i32_kernel<<<(N + 255) / 256, 256, 0, stream>>>(cursor, N);
    degree_kernel<<<(E2 + 255) / 256, 256, 0, stream>>>(edges, E2, E1, cursor);
    exscan_kernel<<<1, 1024, 0, stream>>>(cursor, rowstart, N);
    zero_i32_kernel<<<(N + 255) / 256, 256, 0, stream>>>(cursor, N);
    scatter_kernel<<<(E2 + 255) / 256, 256, 0, stream>>>(edges, E2, E1, rowstart,
                                                         cursor, csr_nbr, csr_eidx);

    node_agg_kernel<<<(N + 7) / 8, 256, 0, stream>>>(rowstart, csr_nbr, csr_eidx,
                                                     scores, qkv, out, N);
  }

  relu_kernel<<<(nS + 255) / 256, 256, 0, stream>>>(out, nS);
}
